// LSTM_60713657696421
// MI455X (gfx1250) — matrix operand order repaired and checked by reading, not compiled, not run on hardware
//
#include <hip/hip_runtime.h>
#include <hip/hip_bf16.h>
#include <math.h>

typedef _Float16 f16;
typedef __attribute__((ext_vector_type(16))) _Float16 v16h;
typedef __attribute__((ext_vector_type(8)))  _Float16 v8h;
typedef __attribute__((ext_vector_type(8)))  float    v8f;

#define H  1024
#define B  16
#define T  512
#define HH (H * H)

// ---- workspace layout (bytes) ----
#define OFF_XH   ((size_t)0)
#define XH_BYTES ((size_t)B * T * H * 2)        // 16 MiB  x as f16
#define OFF_WH   (OFF_XH + XH_BYTES)
#define WH_BYTES ((size_t)4 * HH * 2)           // 8 MiB   Wh[4][H][H] f16
#define OFF_WX   (OFF_WH + WH_BYTES)
#define WX_BYTES ((size_t)4 * HH * 2)           // 8 MiB   Wx[4][H][H] f16
#define OFF_BS   (OFF_WX + WX_BYTES)
#define BS_BYTES ((size_t)4 * H * 4)            // bsum[4][H] = bh+bx, f32
#define OFF_G    (OFF_BS + BS_BYTES)
#define G_BYTES  ((size_t)T * 4 * H * B * 2)    // 64 MiB  G[t][g][j][b] f16 (b contiguous!)
#define OFF_HB   (OFF_G + G_BYTES)
#define HB_BYTES ((size_t)B * H * 2)            // h state f16
#define OFF_BAR  (OFF_HB + HB_BYTES)

#define NWG         64
#define SCAN_THREADS 128
#define WROW        1032   // padded LDS row stride in halves (2064 B)

union FragU { v16h v; v8h h[2]; };
union PackU { v8h v; f16 e[8]; };

// A matrix 16x32 f16: lane<16 -> K 0..7 & 16..23 of row (lane&15); lane>=16 -> +8
__device__ __forceinline__ v16h load_fragA(const f16* __restrict__ p0, int rowstride, int lane) {
  const int r  = lane & 15;
  const int h8 = (lane >> 4) << 3;
  const f16* p = p0 + r * rowstride + h8;
  FragU u;
  u.h[0] = *(const v8h*)(p);
  u.h[1] = *(const v8h*)(p + 16);
  return u.v;
}

// B matrix 32x16 f16: lane holds column (lane&15); lane<16 -> K 0..7 & 16..23, lane>=16 -> +8
__device__ __forceinline__ v16h load_fragB(const f16* __restrict__ p0, int rowstride, int lane) {
  const int n   = lane & 15;
  const int h16 = (lane >> 4) << 3;
  const f16* p  = p0 + n * rowstride + h16;
  FragU u;
  u.h[0] = *(const v8h*)(p);
  u.h[1] = *(const v8h*)(p + 16);
  return u.v;
}

// ---------------- kernel 0: convert + init ----------------
__global__ void lstm_cvt_init(
    const float* __restrict__ x,
    const float* __restrict__ Whf, const float* __restrict__ Whi,
    const float* __restrict__ Whg, const float* __restrict__ Who,
    const float* __restrict__ Wxf, const float* __restrict__ Wxi,
    const float* __restrict__ Wxg, const float* __restrict__ Wxo,
    const float* __restrict__ bhf, const float* __restrict__ bhi,
    const float* __restrict__ bhg, const float* __restrict__ bho,
    const float* __restrict__ bxf, const float* __restrict__ bxi,
    const float* __restrict__ bxg, const float* __restrict__ bxo,
    f16* __restrict__ xh, f16* __restrict__ wh16, f16* __restrict__ wx16,
    float* __restrict__ bsum, int* __restrict__ bar)
{
  const float* whp[4] = {Whf, Whi, Whg, Who};
  const float* wxp[4] = {Wxf, Wxi, Wxg, Wxo};
  const float* bhp[4] = {bhf, bhi, bhg, bho};
  const float* bxp[4] = {bxf, bxi, bxg, bxo};

  const long total = (long)B * T * H;
  const long stride = (long)gridDim.x * blockDim.x;
  for (long i = (long)blockIdx.x * blockDim.x + threadIdx.x; i < total; i += stride) {
    xh[i] = (f16)x[i];
    if (i < HH) {
      #pragma unroll
      for (int g = 0; g < 4; ++g) {
        wh16[(long)g * HH + i] = (f16)whp[g][i];
        wx16[(long)g * HH + i] = (f16)wxp[g][i];
      }
    }
    if (i < 4 * H) {
      int g = (int)(i >> 10), j = (int)(i & (H - 1));
      bsum[i] = bhp[g][j] + bxp[g][j];
    }
    if (i == 0) bar[0] = 0;
  }
}

// ---------------- kernel 1: G[t][g][j][b] = (x_t @ Wx_g^T + bx + bh) transposed ----------------
__global__ __launch_bounds__(256) void lstm_xgate_gemm(
    const f16* __restrict__ xh, const f16* __restrict__ wx16,
    const float* __restrict__ bsum, f16* __restrict__ G)
{
  const int lane = threadIdx.x & 31;
  const int wv   = threadIdx.x >> 5;      // 0..7
  const int t    = blockIdx.x >> 5;       // 0..511
  const int seg  = blockIdx.x & 31;
  const int tile = seg * 8 + wv;          // 0..255 over (g, jtile)
  const int g    = tile >> 6;
  const int jb   = (tile & 63) << 4;

  const f16* Abase = xh + (long)t * H;                    // row stride T*H (rows = b)
  const f16* Bbase = wx16 + ((long)g * H + jb) * H;       // row stride H   (rows = j)

  v8f acc = {};
  #pragma unroll 4
  for (int k = 0; k < H; k += 32) {
    v16h a = load_fragA(Abase + k, T * H, lane);
    v16h b = load_fragB(Bbase + k, H, lane);
    acc = __builtin_amdgcn_wmma_f32_16x16x32_f16(false, a, false, b, (short)0, acc, false, false);
  }

  // C layout: lane holds 8 consecutive M(=b) values at fixed N(=j) -> one b128 store
  const int n  = lane & 15;
  const int hi = lane >> 4;
  const float bias = bsum[g * H + jb + n];
  PackU pk;
  #pragma unroll
  for (int r = 0; r < 8; ++r) pk.e[r] = (f16)(acc[r] + bias);
  f16* Gt = G + ((long)(t * 4 + g) * H + jb + n) * B + 8 * hi;
  *(v8h*)Gt = pk.v;
}

// ---------------- grid barrier across NWG persistent workgroups ----------------
__device__ __forceinline__ void grid_sync(int* bar, int phase) {
  __threadfence();                 // release this thread's global writes
  __syncthreads();
  if (threadIdx.x == 0) {
    atomicAdd(bar, 1);
    while (__hip_atomic_load(bar, __ATOMIC_ACQUIRE, __HIP_MEMORY_SCOPE_AGENT) < NWG * phase) {
      __builtin_amdgcn_s_sleep(2);
    }
  }
  __syncthreads();
  __threadfence();                 // acquire: invalidate for upcoming reads
}

// ---------------- kernel 2: persistent recurrent scan ----------------
// 64 WGs x 128 threads. WG w owns h-columns [16w, 16w+16). Wave g (0..3) computes
// gate g's 16x16 tile each step with K=1024 from LDS-resident Wh slice.
__global__ __launch_bounds__(SCAN_THREADS) void lstm_scan(
    const f16* __restrict__ wh16, const f16* __restrict__ G,
    f16* __restrict__ hbuf, float* __restrict__ hout, int* __restrict__ bar)
{
  extern __shared__ __align__(16) char smem[];
  f16*   whl  = (f16*)smem;                                 // [4*16][WROW] halves
  float* gact = (float*)(smem + (size_t)4 * 16 * WROW * 2); // [4][256]
  float* cst  = gact + 4 * 256;                             // [256]

  const int tid  = threadIdx.x;
  const int lane = tid & 31;
  const int g    = tid >> 5;       // wave index == gate index
  const int w    = blockIdx.x;     // 0..63
  const int jb   = w << 4;

  // Stage this WG's Wh slice (4 gates x 16 rows x 1024 k) into padded LDS.
  for (int idx = tid; idx < 64 * (H / 8); idx += SCAN_THREADS) {
    int row = idx >> 7;            // 0..63  (gate*16 + jlocal)
    int k8  = idx & 127;
    int gg = row >> 4, jl = row & 15;
    const v8h* src = (const v8h*)(wh16 + ((long)(gg * H + jb + jl)) * H) + k8;
    *(v8h*)(whl + (long)row * WROW + k8 * 8) = *src;
  }
  // c0 = 0, h0 = 0 (this WG's slice)
  for (int e = tid; e < 256; e += SCAN_THREADS) {
    cst[e] = 0.f;
    int b = e >> 4, n = e & 15;
    hbuf[b * H + jb + n] = (f16)0.f;
  }

  int phase = 1;
  grid_sync(bar, phase++);

  const f16* wrow = whl + (long)g * 16 * WROW;
  const int n  = lane & 15;
  const int hi = lane >> 4;

  for (int t = 0; t < T; ++t) {
    // gate pre-activation: h_{t-1} @ Wh_g^T  (16x16 tile, K=1024)
    v8f acc = {};
    #pragma unroll 4
    for (int k = 0; k < H; k += 32) {
      v16h a  = load_fragA(hbuf + k, H, lane);     // global, all 1024 h columns
      v16h bb = load_fragB(wrow + k, WROW, lane);  // LDS-resident weights
      acc = __builtin_amdgcn_wmma_f32_16x16x32_f16(false, a, false, bb, (short)0, acc, false, false);
    }

    // + precomputed x-side + biases (one 16B load: 8 consecutive b at fixed j)
    const v8h gx = *(const v8h*)(G + ((long)(t * 4 + g) * H + jb + n) * B + 8 * hi);
    if (t + 1 < T) {  // pull next step's slice toward the WGP during pointwise+barrier
      __builtin_prefetch(G + ((long)((t + 1) * 4 + g) * H + jb + n) * B + 8 * hi, 0, 1);
    }
    #pragma unroll
    for (int r = 0; r < 8; ++r) {
      int m = r + 8 * hi;
      float pre = acc[r] + (float)gx[r];
      float av  = (g == 2) ? tanhf(pre) : 1.f / (1.f + __expf(-pre));
      gact[g * 256 + m * 16 + n] = av;
    }
    __syncthreads();

    // pointwise cell/hidden update for this WG's 16b x 16j slice
    for (int e = tid; e < 256; e += SCAN_THREADS) {
      float fv = gact[0 * 256 + e];
      float iv = gact[1 * 256 + e];
      float gv = gact[2 * 256 + e];
      float ov = gact[3 * 256 + e];
      float c  = cst[e] * fv + iv * gv;
      cst[e] = c;
      float h = tanhf(c) * ov;
      int b = e >> 4, nn = e & 15;
      hbuf[b * H + jb + nn] = (f16)h;
      if (t == T - 1) hout[b * H + jb + nn] = h;   // final output, fp32
    }
    grid_sync(bar, phase++);
  }
}

extern "C" void kernel_launch(void* const* d_in, const int* in_sizes, int n_in,
                              void* d_out, int out_size, void* d_ws, size_t ws_size,
                              hipStream_t stream) {
  (void)in_sizes; (void)n_in; (void)out_size; (void)ws_size;
  // input order: x, Whf,bhf, Whi,bhi, Whg,bhg, Who,bho, Wxf,bxf, Wxi,bxi, Wxg,bxg, Wxo,bxo
  const float* x   = (const float*)d_in[0];
  const float* Whf = (const float*)d_in[1];  const float* bhf = (const float*)d_in[2];
  const float* Whi = (const float*)d_in[3];  const float* bhi = (const float*)d_in[4];
  const float* Whg = (const float*)d_in[5];  const float* bhg = (const float*)d_in[6];
  const float* Who = (const float*)d_in[7];  const float* bho = (const float*)d_in[8];
  const float* Wxf = (const float*)d_in[9];  const float* bxf = (const float*)d_in[10];
  const float* Wxi = (const float*)d_in[11]; const float* bxi = (const float*)d_in[12];
  const float* Wxg = (const float*)d_in[13]; const float* bxg = (const float*)d_in[14];
  const float* Wxo = (const float*)d_in[15]; const float* bxo = (const float*)d_in[16];

  char* ws = (char*)d_ws;
  f16*   xh   = (f16*)(ws + OFF_XH);
  f16*   wh16 = (f16*)(ws + OFF_WH);
  f16*   wx16 = (f16*)(ws + OFF_WX);
  float* bsum = (float*)(ws + OFF_BS);
  f16*   G    = (f16*)(ws + OFF_G);
  f16*   hbuf = (f16*)(ws + OFF_HB);
  int*   bar  = (int*)(ws + OFF_BAR);

  lstm_cvt_init<<<2048, 256, 0, stream>>>(
      x, Whf, Whi, Whg, Who, Wxf, Wxi, Wxg, Wxo,
      bhf, bhi, bhg, bho, bxf, bxi, bxg, bxo,
      xh, wh16, wx16, bsum, bar);

  lstm_xgate_gemm<<<T * 32, 256, 0, stream>>>(xh, wx16, bsum, G);

  const size_t smem = (size_t)4 * 16 * WROW * 2 + 4 * 256 * 4 + 256 * 4; // ~134 KiB
  hipFuncSetAttribute(reinterpret_cast<const void*>(lstm_scan),
                      hipFuncAttributeMaxDynamicSharedMemorySize, (int)smem);
  lstm_scan<<<NWG, SCAN_THREADS, smem, stream>>>(wh16, G, hbuf, (float*)d_out, bar);
}
